// CausalSelfAttention_66228395704450
// MI455X (gfx1250) — matrix operand order repaired and checked by reading, not compiled, not run on hardware
//
#include <hip/hip_runtime.h>
#include <stdint.h>

// ---------------------------------------------------------------------------
// Causal self-attention forward, MI455X (gfx1250, wave32, WMMA).
//
// Pipeline:
//   1. cvt: x, W_kqv (transposed to [h][e][d]), W_proj  -> bf16
//   2. kqv_gemm: per (b,h): X[N,1024] @ WkqvT -> K,Q row-major bf16, V transposed
//      bf16. LDS tiles double-buffered via GLOBAL_LOAD_ASYNC_TO_LDS (ASYNCcnt).
//   3. attn: flash-style online softmax, roles swapped per reference
//      (scores[n,m] = k[n]·q[m], mask m<=n) via v_wmma_f32_16x16x32_bf16
//   4. proj_gemm: sa[BN,1024] @ W_proj^T + b_proj -> fp32 out (same async
//      double-buffered staging)
// ---------------------------------------------------------------------------

typedef __bf16 bf16;
typedef bf16  v16bf __attribute__((ext_vector_type(16)));
typedef bf16  v8bf  __attribute__((ext_vector_type(8)));
typedef bf16  v4bf  __attribute__((ext_vector_type(4)));
typedef float v8f   __attribute__((ext_vector_type(8)));

// Types for the async-to-LDS builtins (from the toolchain's own diagnostics):
// b128 wants v4i pointers, b64 wants v2i pointers; AS1 source, AS3 dest.
typedef int vi4_ __attribute__((vector_size(4 * sizeof(int))));
typedef int vi2_ __attribute__((vector_size(2 * sizeof(int))));
#define AS1_ __attribute__((address_space(1)))
#define AS3_ __attribute__((address_space(3)))

#define B_   2
#define N_   2048
#define D_   1024
#define H_   16
#define HD_  64
#define E3_  192
#define NEG_LARGE (-1.0e30f)

__device__ __forceinline__ v8f wmma_bf16(v16bf a, v16bf b, v8f c) {
  // (neg_a, A, neg_b, B, c_mod, C, reuse_a, reuse_b)
  return __builtin_amdgcn_wmma_f32_16x16x32_bf16(false, a, false, b, (short)0, c,
                                                 false, false);
}

// ------------------- CDNA5 async global->LDS copy helpers -------------------
// LDS operand is the low 32 bits of the flat address
// AS3 pointers are 32-bit.

__device__ __forceinline__ void async_load_b128(const bf16* g, bf16* lds) {
#if __has_builtin(__builtin_amdgcn_global_load_async_to_lds_b128)
  __builtin_amdgcn_global_load_async_to_lds_b128(
      (AS1_ vi4_*)(uintptr_t)g,
      (AS3_ vi4_*)(unsigned)(uintptr_t)lds, 0, 0);
#else
  unsigned l = (unsigned)(uintptr_t)lds;
  asm volatile("global_load_async_to_lds_b128 %0, %1, off"
               :: "v"(l), "v"(g) : "memory");
#endif
}

__device__ __forceinline__ void async_load_b64(const bf16* g, bf16* lds) {
#if __has_builtin(__builtin_amdgcn_global_load_async_to_lds_b64)
  __builtin_amdgcn_global_load_async_to_lds_b64(
      (AS1_ vi2_*)(uintptr_t)g,
      (AS3_ vi2_*)(unsigned)(uintptr_t)lds, 0, 0);
#else
  unsigned l = (unsigned)(uintptr_t)lds;
  asm volatile("global_load_async_to_lds_b64 %0, %1, off"
               :: "v"(l), "v"(g) : "memory");
#endif
}

template <int CNT>
__device__ __forceinline__ void s_wait_async() {
#if __has_builtin(__builtin_amdgcn_s_wait_asynccnt)
  __builtin_amdgcn_s_wait_asynccnt(CNT);
#else
  asm volatile("s_wait_asynccnt %0" :: "i"(CNT) : "memory");
#endif
}

// ------------------------------ WMMA fragments ------------------------------

// B-fragment: per-lane col = lane&15, element idx -> K = 16*half + idx.
// Caller passes pointer already offset to (col row base + 16*half); 16
// contiguous bf16, 16B-aligned.
__device__ __forceinline__ v16bf ld16(const bf16* p) {
  v8bf lo = *(const v8bf*)p;
  v8bf hi = *(const v8bf*)(p + 8);
  v16bf r;
#pragma unroll
  for (int i = 0; i < 8; ++i) { r[i] = lo[i]; r[i + 8] = hi[i]; }
  return r;
}

// A-fragment: per-lane row = lane&15 (caller offsets row base), half = lane>>4.
// idx 0..15 -> K = 16*half + 0..15
__device__ __forceinline__ v16bf ldA(const bf16* rowp, int half) {
  v8bf lo = *(const v8bf*)(rowp + 16 * half);
  v8bf hi = *(const v8bf*)(rowp + 16 * half + 8);
  v16bf r;
#pragma unroll
  for (int i = 0; i < 8; ++i) { r[i] = lo[i]; r[i + 8] = hi[i]; }
  return r;
}

// Row-wise reductions: a D-tile row lives in one 16-lane half-wave.
__device__ __forceinline__ float rowmax16(float v) {
#pragma unroll
  for (int m = 1; m < 16; m <<= 1) v = fmaxf(v, __shfl_xor(v, m, 32));
  return v;
}
__device__ __forceinline__ float rowsum16(float v) {
#pragma unroll
  for (int m = 1; m < 16; m <<= 1) v += __shfl_xor(v, m, 32);
  return v;
}

// --------------------------- conversion kernels ----------------------------

__global__ __launch_bounds__(256) void cvt_bf16_kernel(const float* __restrict__ in,
                                                       bf16* __restrict__ out, int n) {
  int i = blockIdx.x * blockDim.x + threadIdx.x;
  if (i < n) out[i] = (bf16)in[i];
}

// W_kqv [H][D][192] fp32 -> wkqvT [H][192][D] bf16 (so B-fragments read contiguous K)
__global__ __launch_bounds__(256) void cvt_wkqvT_kernel(const float* __restrict__ w,
                                                        bf16* __restrict__ out) {
  int i = blockIdx.x * blockDim.x + threadIdx.x;
  if (i >= H_ * D_ * E3_) return;
  int h = i / (D_ * E3_);
  int rem = i - h * (D_ * E3_);
  int d = rem / E3_;
  int e = rem - d * E3_;
  out[((size_t)h * E3_ + e) * D_ + d] = (bf16)w[i];
}

// ------------------------------- KQV GEMM ----------------------------------
// grid: (B*H, N/32, 3); block 256 (8 waves in a 2Mx4E tile grid).
// C tile per wave: 16x16; K loop over D in steps of 32 through double-buffered
// LDS filled by async global->LDS copies (2 async ops per wave per stage).
__global__ __launch_bounds__(256) void kqv_gemm_kernel(
    const bf16* __restrict__ xb, const bf16* __restrict__ wkqvT,
    const float* __restrict__ bkqv,
    bf16* __restrict__ Kp, bf16* __restrict__ Qp, bf16* __restrict__ Vt) {
  __shared__ bf16 As[2][32 * 40];  // 32 rows (n) x 32 K, stride 40
  __shared__ bf16 Bs[2][64 * 40];  // 64 rows (e) x 32 K, stride 40

  const int bh = blockIdx.x;
  const int n0 = blockIdx.y * 32;
  const int eblk = blockIdx.z;
  const int e0 = eblk * 64;
  const int h = bh & 15;
  const int t = threadIdx.x;
  const int wave = t >> 5, lane = t & 31;
  const int wm = wave >> 2, we = wave & 3;
  const int half = lane >> 4, lrow = lane & 15;

  const int b = bh >> 4;
  const bf16* xrow = xb + ((size_t)b * N_ + n0) * D_;
  const bf16* wrow = wkqvT + ((size_t)h * E3_ + e0) * D_;

  const int ar = t >> 3, ac = (t & 7) * 4;  // A tile: 32 rows x 4 bf16 (b64/thread)
  const int br = t >> 2, bc = (t & 3) * 8;  // B tile: 64 rows x 8 bf16 (b128/thread)

  const bf16* ag = xrow + (size_t)ar * D_ + ac;
  const bf16* bg = wrow + (size_t)br * D_ + bc;
  bf16* aw[2] = {&As[0][ar * 40 + ac], &As[1][ar * 40 + ac]};
  bf16* bw[2] = {&Bs[0][br * 40 + bc], &Bs[1][br * 40 + bc]};

  // prologue: stage 0 in flight
  async_load_b64(ag, aw[0]);
  async_load_b128(bg, bw[0]);

  v8f acc = {};
#pragma unroll 4
  for (int k0 = 0; k0 < D_; k0 += 32) {
    const int buf = (k0 >> 5) & 1;
    if (k0 + 32 < D_) {
      async_load_b64(ag + k0 + 32, aw[buf ^ 1]);
      async_load_b128(bg + k0 + 32, bw[buf ^ 1]);
      s_wait_async<2>();   // stage k0 landed (next stage still in flight)
    } else {
      s_wait_async<0>();
    }
    __syncthreads();       // publish this stage's LDS to all waves
    v16bf a  = ldA(&As[buf][(wm * 16 + lrow) * 40], half);
    v16bf bb = ld16(&Bs[buf][(we * 16 + lrow) * 40 + 16 * half]);
    acc = wmma_bf16(a, bb, acc);
    __syncthreads();       // everyone done reading buf before it is refilled
  }

  const int col = we * 16 + lrow;            // local e within 64-wide block
  const float bias = bkqv[h * E3_ + e0 + col];

  if (eblk == 2) {
    // V stored transposed: Vt[bh][hd][n]; lane's 8 rows are contiguous in n.
    bf16* dst = Vt + ((size_t)bh * HD_ + col) * N_ + n0 + wm * 16 + 8 * half;
    v8bf pk;
#pragma unroll
    for (int r = 0; r < 8; ++r) pk[r] = (bf16)(acc[r] + bias);
    *(v8bf*)dst = pk;
  } else {
    bf16* base = (eblk == 0) ? Kp : Qp;      // chunk order in reference: k, q, v
    size_t row0 = (size_t)bh * N_ + n0 + wm * 16 + 8 * half;
#pragma unroll
    for (int r = 0; r < 8; ++r) base[(row0 + r) * HD_ + col] = (bf16)(acc[r] + bias);
  }
}

// ----------------------------- flash attention -----------------------------
// Reference computes scores = K @ Q^T with causal tril mask, softmax over m,
// then attn @ V. So rows <- Kproj, cols <- Qproj/V. One wave per 16-row tile.
__global__ __launch_bounds__(128) void attn_kernel(
    const bf16* __restrict__ Kp, const bf16* __restrict__ Qp,
    const bf16* __restrict__ Vt, bf16* __restrict__ sab) {
  __shared__ bf16 pbuf[4][16 * 40];  // per-wave P transpose buffer (16x32, pad 40)

  const int warp = threadIdx.x >> 5;
  const int lane = threadIdx.x & 31;
  const int gid  = blockIdx.x * 4 + warp;
  const int tile = gid & 127;        // N/16 = 128 row tiles
  const int bh   = gid >> 7;         // 0..31
  const int b    = bh >> 4;
  const int h    = bh & 15;
  const int i0   = tile * 16;
  const int half = lane >> 4;
  const int lrow = lane & 15;

  const bf16* Kb = Kp + (size_t)bh * N_ * HD_;
  const bf16* Qb = Qp + (size_t)bh * N_ * HD_;
  const bf16* Vb = Vt + (size_t)bh * HD_ * N_;

  // A fragments (row block of Kproj) reused across all column blocks.
  v16bf a0 = ldA(Kb + (size_t)(i0 + lrow) * HD_ + 0,  half);
  v16bf a1 = ldA(Kb + (size_t)(i0 + lrow) * HD_ + 32, half);

  float m[8], l[8];
  v8f O[4] = {};
#pragma unroll
  for (int r = 0; r < 8; ++r) { m[r] = NEG_LARGE; l[r] = 0.0f; }

  const int jbmax = (i0 + 15) >> 5;  // causal: only blocks with some j <= i
  for (int jb = 0; jb <= jbmax; ++jb) {
    const int j0 = jb * 32;
    v8f s0, s1;
    {
      const bf16* q = Qb + (size_t)(j0 + lrow) * HD_;
      v8f c = {};
      c = wmma_bf16(a0, ld16(q + 16 * half), c);
      c = wmma_bf16(a1, ld16(q + 32 + 16 * half), c);
      s0 = c;
    }
    {
      const bf16* q = Qb + (size_t)(j0 + 16 + lrow) * HD_;
      v8f c = {};
      c = wmma_bf16(a0, ld16(q + 16 * half), c);
      c = wmma_bf16(a1, ld16(q + 32 + 16 * half), c);
      s1 = c;
    }

    float alpha[8];
#pragma unroll
    for (int r = 0; r < 8; ++r) {
      const int i = i0 + r + 8 * half;
      float v0 = s0[r] * 0.125f;   // 1/sqrt(64)
      float v1 = s1[r] * 0.125f;
      s0[r] = ((j0 + lrow) <= i) ? v0 : NEG_LARGE;
      s1[r] = ((j0 + 16 + lrow) <= i) ? v1 : NEG_LARGE;
      float mx = rowmax16(fmaxf(s0[r], s1[r]));
      float mn = fmaxf(m[r], mx);
      alpha[r] = __expf(m[r] - mn);
      m[r] = mn;
      float p0 = __expf(s0[r] - mn);
      float p1 = __expf(s1[r] - mn);
      // D-layout -> LDS so we can re-read as an A fragment (bf16).
      pbuf[warp][(r + 8 * half) * 40 + lrow]      = (bf16)p0;
      pbuf[warp][(r + 8 * half) * 40 + 16 + lrow] = (bf16)p1;
      l[r] = l[r] * alpha[r] + rowsum16(p0 + p1);
    }

#pragma unroll
    for (int ht = 0; ht < 4; ++ht)
#pragma unroll
      for (int r = 0; r < 8; ++r) O[ht][r] *= alpha[r];

    // Same-wave LDS RAW: compiler inserts s_wait_dscnt. No block barrier
    // (waves have different trip counts).
    v16bf pA = ldA(&pbuf[warp][lrow * 40], half);
#pragma unroll
    for (int ht = 0; ht < 4; ++ht) {
      v16bf bv = ld16(Vb + (size_t)(ht * 16 + lrow) * N_ + j0 + 16 * half);
      O[ht] = wmma_bf16(pA, bv, O[ht]);
    }
  }

#pragma unroll
  for (int r = 0; r < 8; ++r) l[r] = 1.0f / l[r];
  // concat heads: sa[b][n][h*HD + hd]
  bf16* outp = sab + ((size_t)b * N_ + i0) * D_ + h * HD_;
#pragma unroll
  for (int ht = 0; ht < 4; ++ht)
#pragma unroll
    for (int r = 0; r < 8; ++r)
      outp[(size_t)(r + 8 * half) * D_ + ht * 16 + lrow] = (bf16)(O[ht][r] * l[r]);
}

// ---------------------------- output projection ----------------------------
// out[bn][j] = sum_d sa[bn][d] * W_proj[j][d] + b_proj[j]
// grid: (BN/32, D/64); block 256, same async double-buffered staging.
__global__ __launch_bounds__(256) void proj_gemm_kernel(
    const bf16* __restrict__ sab, const bf16* __restrict__ wpb,
    const float* __restrict__ bproj, float* __restrict__ out) {
  __shared__ bf16 As[2][32 * 40];
  __shared__ bf16 Bs[2][64 * 40];

  const int n0 = blockIdx.x * 32;
  const int j0 = blockIdx.y * 64;
  const int t = threadIdx.x;
  const int wave = t >> 5, lane = t & 31;
  const int wm = wave >> 2, we = wave & 3;
  const int half = lane >> 4, lrow = lane & 15;

  const bf16* arow = sab + (size_t)n0 * D_;
  const bf16* wrow = wpb + (size_t)j0 * D_;  // W_proj row-major [j][d] == B^T layout

  const int ar = t >> 3, ac = (t & 7) * 4;
  const int br = t >> 2, bc = (t & 3) * 8;

  const bf16* ag = arow + (size_t)ar * D_ + ac;
  const bf16* bg = wrow + (size_t)br * D_ + bc;
  bf16* aw[2] = {&As[0][ar * 40 + ac], &As[1][ar * 40 + ac]};
  bf16* bw[2] = {&Bs[0][br * 40 + bc], &Bs[1][br * 40 + bc]};

  async_load_b64(ag, aw[0]);
  async_load_b128(bg, bw[0]);

  v8f acc = {};
#pragma unroll 4
  for (int k0 = 0; k0 < D_; k0 += 32) {
    const int buf = (k0 >> 5) & 1;
    if (k0 + 32 < D_) {
      async_load_b64(ag + k0 + 32, aw[buf ^ 1]);
      async_load_b128(bg + k0 + 32, bw[buf ^ 1]);
      s_wait_async<2>();
    } else {
      s_wait_async<0>();
    }
    __syncthreads();
    v16bf a  = ldA(&As[buf][(wm * 16 + lrow) * 40], half);
    v16bf bb = ld16(&Bs[buf][(we * 16 + lrow) * 40 + 16 * half]);
    acc = wmma_bf16(a, bb, acc);
    __syncthreads();
  }

  const int col = we * 16 + lrow;
  const float bias = bproj[j0 + col];
#pragma unroll
  for (int r = 0; r < 8; ++r)
    out[(size_t)(n0 + wm * 16 + 8 * half + r) * D_ + j0 + col] = acc[r] + bias;
}

// --------------------------------- launch ----------------------------------

extern "C" void kernel_launch(void* const* d_in, const int* in_sizes, int n_in,
                              void* d_out, int out_size, void* d_ws, size_t ws_size,
                              hipStream_t stream) {
  (void)in_sizes; (void)n_in; (void)out_size; (void)ws_size;
  const float* x     = (const float*)d_in[0];
  const float* Wkqv  = (const float*)d_in[1];
  const float* bkqv  = (const float*)d_in[2];
  const float* Wproj = (const float*)d_in[3];
  const float* bproj = (const float*)d_in[4];
  float* out = (float*)d_out;

  char* p = (char*)d_ws;
  bf16* xb    = (bf16*)p; p += (size_t)B_ * N_ * D_ * 2;       // 8 MB
  bf16* wkqvT = (bf16*)p; p += (size_t)H_ * E3_ * D_ * 2;      // 6 MB
  bf16* wpb   = (bf16*)p; p += (size_t)D_ * D_ * 2;            // 2 MB
  bf16* Kp    = (bf16*)p; p += (size_t)B_ * H_ * N_ * HD_ * 2; // 8 MB
  bf16* Qp    = (bf16*)p; p += (size_t)B_ * H_ * N_ * HD_ * 2; // 8 MB
  bf16* Vt    = (bf16*)p; p += (size_t)B_ * H_ * N_ * HD_ * 2; // 8 MB
  bf16* sab   = xb;  // xb is fully consumed by kqv_gemm before attn writes sa

  cvt_bf16_kernel<<<(B_ * N_ * D_) / 256, 256, 0, stream>>>(x, xb, B_ * N_ * D_);
  cvt_wkqvT_kernel<<<(H_ * D_ * E3_) / 256, 256, 0, stream>>>(Wkqv, wkqvT);
  cvt_bf16_kernel<<<(D_ * D_) / 256, 256, 0, stream>>>(Wproj, wpb, D_ * D_);

  kqv_gemm_kernel<<<dim3(B_ * H_, N_ / 32, 3), 256, 0, stream>>>(xb, wkqvT, bkqv,
                                                                 Kp, Qp, Vt);
  attn_kernel<<<(B_ * H_ * (N_ / 16)) / 4, 128, 0, stream>>>(Kp, Qp, Vt, sab);
  proj_gemm_kernel<<<dim3((B_ * N_) / 32, D_ / 64), 256, 0, stream>>>(sab, wpb,
                                                                      bproj, out);
}
